// MultiHead_81527069213242
// MI455X (gfx1250) — matrix operand order repaired and checked by reading, not compiled, not run on hardware
//
#include <hip/hip_runtime.h>
#include <hip/hip_bf16.h>
#include <math.h>
#include <stdint.h>

// Problem constants (from reference)
#define EMB   1024
#define NSEQ  4096
#define SCL   0.125f        // 1/sqrt(64)
#define MASKV (-1e9f)

typedef __attribute__((ext_vector_type(16))) _Float16 v16h;
typedef __attribute__((ext_vector_type(8)))  _Float16 v8h;
typedef __attribute__((ext_vector_type(8)))  float    v8f;

__device__ __forceinline__ v8f wmma16(v16h a, v16h b, v8f c) {
  // D = A(16x32 f16) * B(32x16 f16) + C(16x16 f32)
  return __builtin_amdgcn_wmma_f32_16x16x32_f16(false, a, false, b, (short)0, c,
                                                false, false);
}

// ---- CDNA5 async global->LDS copy (ASYNCcnt-tracked, ISA ch.10 async ops) ----
__device__ __forceinline__ void async_copy16(uint32_t lds_byte_addr,
                                             const void* gptr) {
  asm volatile("global_load_async_to_lds_b128 %0, %1, off"
               :
               : "v"(lds_byte_addr), "v"(gptr)
               : "memory");
}
__device__ __forceinline__ void wait_async0() {
  asm volatile("s_wait_asynccnt 0" ::: "memory");
}
__device__ __forceinline__ uint32_t lds_addr_of(const void* p) {
  // LDS aperture: addr[31:0] is the wave-relative LDS byte offset
  return (uint32_t)(uintptr_t)p;
}

// A fragment: 16x32 tile of row-major f16 matrix (ld = row stride in halves).
// ISA layout: lane L holds M = m0+(L&15); halves 0..7 -> K = k0+8h+0..7,
// halves 8..15 -> K = k0+16+8h+0..7  (h = L>>4).
__device__ __forceinline__ v16h frag_a(const _Float16* base, int ld, int m0,
                                       int k0, int lane) {
  const int m = m0 + (lane & 15);
  const int h = lane >> 4;
  const _Float16* p = base + (size_t)m * ld + k0 + h * 8;
  v8h lo = *(const v8h*)(p);
  v8h hi = *(const v8h*)(p + 16);
  return __builtin_shufflevector(lo, hi, 0, 1, 2, 3, 4, 5, 6, 7, 8, 9, 10, 11,
                                 12, 13, 14, 15);
}

// B fragment from a TRANSPOSED operand Bt[n, k] (so B[k,n] = Bt[n,k]).
// ISA layout: lane L holds N = n0+(L&15); halves 0..7 -> K = k0+8h+0..7,
// halves 8..15 -> K = k0+16+8h+0..7  (h = L>>4).
__device__ __forceinline__ v16h frag_bt(const _Float16* baseT, int ld, int n0,
                                        int k0, int lane) {
  const int n = n0 + (lane & 15);
  const int h = lane >> 4;
  const _Float16* p = baseT + (size_t)n * ld + k0 + h * 8;
  v8h lo = *(const v8h*)(p);
  v8h hi = *(const v8h*)(p + 16);
  return __builtin_shufflevector(lo, hi, 0, 1, 2, 3, 4, 5, 6, 7, 8, 9, 10, 11,
                                 12, 13, 14, 15);
}

// ---------------- conversion kernels ----------------
__global__ void k_cvt_f16(const float* __restrict__ src,
                          _Float16* __restrict__ dst, int n) {
  int i = blockIdx.x * blockDim.x + threadIdx.x;
  if (i < n) dst[i] = (_Float16)src[i];
}

// W[rows, cols] (f32) -> Wt[cols, rows] (f16)
__global__ void k_transpose_cvt(const float* __restrict__ W,
                                _Float16* __restrict__ Wt, int rows, int cols) {
  int i = blockIdx.x * blockDim.x + threadIdx.x;
  if (i >= rows * cols) return;
  int r = i / cols, c = i % cols;
  Wt[(size_t)c * rows + r] = (_Float16)W[i];
}

// Stage a 128x32-half tile (row-major, ld halves) into LDS via async copies.
// 512 16-byte chunks spread over 256 threads (2 each).
__device__ __forceinline__ void stage_tile128x32(const _Float16* __restrict__ g,
                                                 int ld, int row0, int k0,
                                                 uint32_t lds_base, int tid) {
#pragma unroll
  for (int i = 0; i < 2; ++i) {
    const int ch = tid + 256 * i;          // ch = row*4 + seg
    const int row = ch >> 2, seg = ch & 3;
    const _Float16* p = g + (size_t)(row0 + row) * ld + k0 + seg * 8;
    async_copy16(lds_base + (uint32_t)(ch * 16), p);
  }
}

// ---------------- QKV projection GEMM ----------------
// C[M=4096, N=1024] = Xh @ W + b, W given transposed (Wt[n, k]).
// Block tile 128x128, 8 waves as 2(M)x4(N), wave tile 64x32.
// A and B tiles double-buffered in LDS via async global->LDS copies.
template <bool TR>
__global__ __launch_bounds__(256) void k_gemm_qkv(
    const _Float16* __restrict__ Xh, const _Float16* __restrict__ Wt,
    const float* __restrict__ bias, _Float16* __restrict__ out) {
  __shared__ _Float16 As[2][128 * 32];  // 2 x 8 KB
  __shared__ _Float16 Bs[2][128 * 32];  // 2 x 8 KB
  const int tid = threadIdx.x;
  const int lane = tid & 31, w = tid >> 5;
  const int wm = w >> 2, wn = w & 3;
  const int m0b = blockIdx.x * 128, n0b = blockIdx.y * 128;
  const int m0l = wm * 64, n0l = wn * 32;

  const uint32_t asb[2] = {lds_addr_of(&As[0][0]), lds_addr_of(&As[1][0])};
  const uint32_t bsb[2] = {lds_addr_of(&Bs[0][0]), lds_addr_of(&Bs[1][0])};

  // prologue: stage K-tile 0
  stage_tile128x32(Xh, EMB, m0b, 0, asb[0], tid);
  stage_tile128x32(Wt, EMB, n0b, 0, bsb[0], tid);

  v8f acc[4][2] = {};
#pragma unroll 1
  for (int kt = 0; kt < EMB / 32; ++kt) {
    wait_async0();       // our async writes of buf[kt&1] have landed
    __syncthreads();     // everyone's have landed; prior reads of next buf done
    if (kt + 1 < EMB / 32) {
      stage_tile128x32(Xh, EMB, m0b, (kt + 1) * 32, asb[(kt + 1) & 1], tid);
      stage_tile128x32(Wt, EMB, n0b, (kt + 1) * 32, bsb[(kt + 1) & 1], tid);
    }
    const _Float16* A = &As[kt & 1][0];
    const _Float16* B = &Bs[kt & 1][0];
    v16h a[4], b[2];
#pragma unroll
    for (int i = 0; i < 4; ++i) a[i] = frag_a(A, 32, m0l + 16 * i, 0, lane);
#pragma unroll
    for (int j = 0; j < 2; ++j) b[j] = frag_bt(B, 32, n0l + 16 * j, 0, lane);
#pragma unroll
    for (int i = 0; i < 4; ++i)
#pragma unroll
      for (int j = 0; j < 2; ++j) acc[i][j] = wmma16(a[i], b[j], acc[i][j]);
  }

  const int c = lane & 15, h = lane >> 4;
#pragma unroll
  for (int i = 0; i < 4; ++i)
#pragma unroll
    for (int j = 0; j < 2; ++j) {
      const int col = n0b + n0l + 16 * j + c;
      const float bcol = bias[col];
      if (TR) {
        // vT[n, m]: 8 consecutive rows at fixed col -> one b128 store
        v8h pk;
#pragma unroll
        for (int r = 0; r < 8; ++r) pk[r] = (_Float16)(acc[i][j][r] + bcol);
        *(v8h*)(out + (size_t)col * NSEQ + m0b + m0l + 16 * i + 8 * h) = pk;
      } else {
#pragma unroll
        for (int r = 0; r < 8; ++r) {
          const int row = m0b + m0l + 16 * i + 8 * h + r;
          out[(size_t)row * EMB + col] = (_Float16)(acc[i][j][r] + bcol);
        }
      }
    }
}

// Stage the block's contiguous 32x1024 q tile (64 KB) into LDS.
__device__ __forceinline__ void stage_qtile(const _Float16* __restrict__ q,
                                            int row0, uint32_t lds_base,
                                            int tid) {
  const _Float16* g = q + (size_t)row0 * EMB;  // contiguous 64 KB block
#pragma unroll
  for (int i = 0; i < 16; ++i) {
    const int ch = tid + 256 * i;  // 4096 x 16B chunks
    async_copy16(lds_base + (uint32_t)(ch * 16), g + ch * 8);
  }
}

// ---------------- pass 1: per-row softmax stats ----------------
// Block: 32 query rows; wave w owns cols [w*512, w*512+512).
__global__ __launch_bounds__(256) void k_attn_stats(
    const _Float16* __restrict__ q, const _Float16* __restrict__ k,
    float* __restrict__ rowM, float* __restrict__ rowL) {
  __shared__ _Float16 qs[32 * EMB];  // 64 KB staged q tile
  __shared__ float smax[8][32];
  __shared__ float ssum[8][32];
  const int tid = threadIdx.x;
  const int lane = tid & 31, w = tid >> 5;
  const int c = lane & 15, h = lane >> 4;
  const int row0 = blockIdx.x * 32;

  stage_qtile(q, row0, lds_addr_of(&qs[0]), tid);
  wait_async0();
  __syncthreads();

  float m[16], l[16];
#pragma unroll
  for (int i = 0; i < 16; ++i) { m[i] = -INFINITY; l[i] = 0.0f; }

#pragma unroll 1
  for (int ct = 0; ct < 16; ++ct) {
    const int col0 = w * 512 + ct * 32;
    v8f acc[2][2] = {};
#pragma unroll 1
    for (int k0 = 0; k0 < EMB; k0 += 32) {
      v16h a[2], b[2];
#pragma unroll
      for (int mi = 0; mi < 2; ++mi)
        a[mi] = frag_a(qs, EMB, 16 * mi, k0, lane);
#pragma unroll
      for (int ni = 0; ni < 2; ++ni)
        b[ni] = frag_bt(k, EMB, col0 + 16 * ni, k0, lane);
#pragma unroll
      for (int mi = 0; mi < 2; ++mi)
#pragma unroll
        for (int ni = 0; ni < 2; ++ni)
          acc[mi][ni] = wmma16(a[mi], b[ni], acc[mi][ni]);
    }
    // scale + inverted causal mask + online (max,sum) per row
#pragma unroll
    for (int mi = 0; mi < 2; ++mi)
#pragma unroll
      for (int r = 0; r < 8; ++r) {
        const int idx = mi * 8 + r;
        const int row = row0 + mi * 16 + 8 * h + r;
        float v0 = acc[mi][0][r] * SCL;
        float v1 = acc[mi][1][r] * SCL;
        if (col0 + c <= row) v0 = MASKV;
        if (col0 + 16 + c <= row) v1 = MASKV;
        float tmax = fmaxf(v0, v1);
        for (int off = 8; off > 0; off >>= 1)
          tmax = fmaxf(tmax, __shfl_xor(tmax, off, 16));
        const float nm = fmaxf(m[idx], tmax);
        float e = __expf(v0 - nm) + __expf(v1 - nm);
        for (int off = 8; off > 0; off >>= 1) e += __shfl_xor(e, off, 16);
        l[idx] = l[idx] * __expf(m[idx] - nm) + e;
        m[idx] = nm;
      }
  }

  if (c == 0) {
#pragma unroll
    for (int mi = 0; mi < 2; ++mi)
#pragma unroll
      for (int r = 0; r < 8; ++r) {
        const int rl = mi * 16 + 8 * h + r;
        smax[w][rl] = m[mi * 8 + r];
        ssum[w][rl] = l[mi * 8 + r];
      }
  }
  __syncthreads();
  if (tid < 32) {
    float M = -INFINITY, S = 0.0f;
#pragma unroll
    for (int ww = 0; ww < 8; ++ww) M = fmaxf(M, smax[ww][tid]);
#pragma unroll
    for (int ww = 0; ww < 8; ++ww) S += ssum[ww][tid] * __expf(smax[ww][tid] - M);
    rowM[row0 + tid] = M;
    rowL[row0 + tid] = S;
  }
}

// ---------------- pass 2: out = attn @ v + X ----------------
// Block: 32 rows. Dynamic LDS = 256 KB probability tile P[32][4096] +
// 64 KB staged q tile (320 KB = full CDNA5 WGP LDS).
// Phase 1 recomputes scores, writes normalized probs (f16) into P.
// Phase 2: WMMA P @ vT, + residual.
__global__ __launch_bounds__(256) void k_attn_out(
    const _Float16* __restrict__ q, const _Float16* __restrict__ k,
    const _Float16* __restrict__ vT, const float* __restrict__ rowM,
    const float* __restrict__ rowL, const float* __restrict__ X,
    float* __restrict__ out) {
  extern __shared__ _Float16 smem[];
  _Float16* P = smem;               // [32][NSEQ]
  _Float16* qs = smem + 32 * NSEQ;  // [32][EMB]
  const int tid = threadIdx.x;
  const int lane = tid & 31, w = tid >> 5;
  const int c = lane & 15, h = lane >> 4;
  const int row0 = blockIdx.x * 32;

  stage_qtile(q, row0, lds_addr_of(qs), tid);

  float Mr[16], Li[16];
#pragma unroll
  for (int mi = 0; mi < 2; ++mi)
#pragma unroll
    for (int r = 0; r < 8; ++r) {
      const int row = row0 + mi * 16 + 8 * h + r;
      Mr[mi * 8 + r] = rowM[row];
      Li[mi * 8 + r] = 1.0f / rowL[row];
    }
  wait_async0();
  __syncthreads();

  // ---- phase 1: probabilities into LDS ----
#pragma unroll 1
  for (int ct = 0; ct < 16; ++ct) {
    const int col0 = w * 512 + ct * 32;
    v8f acc[2][2] = {};
#pragma unroll 1
    for (int k0 = 0; k0 < EMB; k0 += 32) {
      v16h a[2], b[2];
#pragma unroll
      for (int mi = 0; mi < 2; ++mi)
        a[mi] = frag_a(qs, EMB, 16 * mi, k0, lane);
#pragma unroll
      for (int ni = 0; ni < 2; ++ni)
        b[ni] = frag_bt(k, EMB, col0 + 16 * ni, k0, lane);
#pragma unroll
      for (int mi = 0; mi < 2; ++mi)
#pragma unroll
        for (int ni = 0; ni < 2; ++ni)
          acc[mi][ni] = wmma16(a[mi], b[ni], acc[mi][ni]);
    }
#pragma unroll
    for (int mi = 0; mi < 2; ++mi)
#pragma unroll
      for (int r = 0; r < 8; ++r) {
        const int idx = mi * 8 + r;
        const int row = row0 + mi * 16 + 8 * h + r;
        const int rl = mi * 16 + 8 * h + r;
        float v0 = acc[mi][0][r] * SCL;
        float v1 = acc[mi][1][r] * SCL;
        if (col0 + c <= row) v0 = MASKV;
        if (col0 + 16 + c <= row) v1 = MASKV;
        P[(size_t)rl * NSEQ + col0 + c]      = (_Float16)(__expf(v0 - Mr[idx]) * Li[idx]);
        P[(size_t)rl * NSEQ + col0 + 16 + c] = (_Float16)(__expf(v1 - Mr[idx]) * Li[idx]);
      }
  }
  __syncthreads();

  // ---- phase 2: out[32, 1024] = P @ v + X ----
  const int cb = w * 128;  // this wave's 128 output columns
#pragma unroll 1
  for (int nt = 0; nt < 8; ++nt) {
    const int n0 = cb + nt * 16;
    v8f acc[2] = {};
#pragma unroll 1
    for (int j0 = 0; j0 < NSEQ; j0 += 32) {
      v16h a0 = frag_a(P, NSEQ, 0, j0, lane);
      v16h a1 = frag_a(P, NSEQ, 16, j0, lane);
      v16h b = frag_bt(vT, NSEQ, n0, j0, lane);
      acc[0] = wmma16(a0, b, acc[0]);
      acc[1] = wmma16(a1, b, acc[1]);
    }
    const int col = n0 + c;
#pragma unroll
    for (int mi = 0; mi < 2; ++mi)
#pragma unroll
      for (int r = 0; r < 8; ++r) {
        const int row = row0 + mi * 16 + 8 * h + r;
        out[(size_t)row * EMB + col] = acc[mi][r] + X[(size_t)row * EMB + col];
      }
  }
}

extern "C" void kernel_launch(void* const* d_in, const int* in_sizes, int n_in,
                              void* d_out, int out_size, void* d_ws,
                              size_t ws_size, hipStream_t stream) {
  const float* X  = (const float*)d_in[0];
  const float* Wq = (const float*)d_in[1];
  const float* bq = (const float*)d_in[2];
  const float* Wk = (const float*)d_in[3];
  const float* bk = (const float*)d_in[4];
  const float* Wv = (const float*)d_in[5];
  const float* bv = (const float*)d_in[6];
  float* out = (float*)d_out;

  char* ws = (char*)d_ws;
  const size_t MB = 1u << 20;
  _Float16* Xh  = (_Float16*)(ws);              // 8 MB  [4096,1024]
  _Float16* qh  = (_Float16*)(ws + 8 * MB);     // 8 MB  [4096,1024]
  _Float16* kh  = (_Float16*)(ws + 16 * MB);    // 8 MB  [4096,1024]
  _Float16* vT  = (_Float16*)(ws + 24 * MB);    // 8 MB  [1024,4096]
  _Float16* Wqt = (_Float16*)(ws + 32 * MB);    // 2 MB  [1024,1024] transposed
  _Float16* Wkt = (_Float16*)(ws + 34 * MB);    // 2 MB
  _Float16* Wvt = (_Float16*)(ws + 36 * MB);    // 2 MB
  float* rowM   = (float*)(ws + 38 * MB);       // 16 KB
  float* rowL   = (float*)(ws + 38 * MB + 16384);

  // convert inputs/weights to f16 (weights transposed for contiguous B frags)
  k_cvt_f16<<<(NSEQ * EMB) / 256, 256, 0, stream>>>(X, Xh, NSEQ * EMB);
  k_transpose_cvt<<<(EMB * EMB) / 256, 256, 0, stream>>>(Wq, Wqt, EMB, EMB);
  k_transpose_cvt<<<(EMB * EMB) / 256, 256, 0, stream>>>(Wk, Wkt, EMB, EMB);
  k_transpose_cvt<<<(EMB * EMB) / 256, 256, 0, stream>>>(Wv, Wvt, EMB, EMB);

  // q, k (row-major) and v (transposed) projections
  dim3 g2(NSEQ / 128, EMB / 128);
  k_gemm_qkv<false><<<g2, 256, 0, stream>>>(Xh, Wqt, bq, qh);
  k_gemm_qkv<false><<<g2, 256, 0, stream>>>(Xh, Wkt, bk, kh);
  k_gemm_qkv<true><<<g2, 256, 0, stream>>>(Xh, Wvt, bv, vT);

  // softmax row stats, then attention output with residual
  k_attn_stats<<<NSEQ / 32, 256, 0, stream>>>(qh, kh, rowM, rowL);
  const size_t lds_bytes = (size_t)(32 * NSEQ + 32 * EMB) * sizeof(_Float16);
  k_attn_out<<<NSEQ / 32, 256, lds_bytes, stream>>>(qh, kh, vT, rowM, rowL, X,
                                                    out);
}
